// Local_Inference_Modeling_74878459838619
// MI455X (gfx1250) — matrix operand order repaired and checked by reading, not compiled, not run on hardware
//
#include <hip/hip_runtime.h>
#include <math.h>

// ---------------------------------------------------------------------------
// Local Inference Modeling block for MI455X (gfx1250, wave32, WMMA).
//   e        = A @ B^T                (bf16 WMMA, fp32 accum, fp32 e in ws)
//   a_tilde  = softmax_row(e) @ B     (bf16 WMMA)
//   b_tilde  = softmax_col(e)^T @ A   (bf16 WMMA)
//   m_a = [A, a~, A-a~, A*a~],  m_b = [B, b~, B-b~, B*b~]  (fused epilogues)
//
// GEMMs: double-buffered LDS, register-staged software pipeline (next tile's
// global_load_b128s issue before the current tile's WMMAs), one barrier per
// K-step. bf16 conversion via native fptrunc (RNE) so the backend can pick
// v_cvt_pk_bf16_f32. Column softmax stats split into partial+combine for
// block-level parallelism.
// ---------------------------------------------------------------------------

typedef __attribute__((ext_vector_type(16))) __bf16          v16bf;
typedef __attribute__((ext_vector_type(8)))  float           v8f;
typedef __attribute__((ext_vector_type(8)))  unsigned short  v8u;
typedef __attribute__((ext_vector_type(16))) unsigned short  v16u;

#define SEQ   512
#define BATCH 32
#define TM    128     // workgroup tile M
#define TN    128     // workgroup tile N
#define TK    32      // K step (one bf16 WMMA worth)
#define PITCH 40      // LDS row pitch in bf16 elems (80 bytes, 16B multiple)
#define NSTEP (SEQ / TK)

// float -> bf16 via native fptrunc (round-to-nearest-even)
__device__ __forceinline__ unsigned short f2bf1(float x) {
    return __builtin_bit_cast(unsigned short, (__bf16)x);
}

// pack 4 floats -> 4 bf16 in a uint2 (one ds_store_b64)
__device__ __forceinline__ uint2 pack4bf(float4 v) {
    uint2 r;
    r.x = (unsigned)f2bf1(v.x) | ((unsigned)f2bf1(v.y) << 16);
    r.y = (unsigned)f2bf1(v.z) | ((unsigned)f2bf1(v.w) << 16);
    return r;
}

// A-operand fragment (16x32 bf16). ISA layout: lane = 16*half + m;
// element e<8 -> K = 8*half + e ; e>=8 -> K = 16 + 8*half + (e-8).
__device__ __forceinline__ v16bf load_frag_a(const unsigned short* lds, int mBase, int lane) {
    int m = mBase + (lane & 15);
    int half = lane >> 4;
    const unsigned short* row = lds + m * PITCH;
    v8u lo = *(const v8u*)(row + 8 * half);
    v8u hi = *(const v8u*)(row + 8 * half + 16);
    v16u t = __builtin_shufflevector(lo, hi, 0,1,2,3,4,5,6,7,8,9,10,11,12,13,14,15);
    return __builtin_bit_cast(v16bf, t);
}

// B-operand fragment (32x16 bf16). ISA layout: lane = 16*half + n;
// element e<8 -> K = 8*half + e ; e>=8 -> K = 16 + 8*half + (e-8).  LDS tile stored as [n][k].
__device__ __forceinline__ v16bf load_frag_b(const unsigned short* lds, int nBase, int lane) {
    int n = nBase + (lane & 15);
    int half = lane >> 4;
    const unsigned short* row = lds + n * PITCH;
    v8u lo = *(const v8u*)(row + 8 * half);
    v8u hi = *(const v8u*)(row + 8 * half + 16);
    v16u t = __builtin_shufflevector(lo, hi, 0,1,2,3,4,5,6,7,8,9,10,11,12,13,14,15);
    return __builtin_bit_cast(v16bf, t);
}

#define WMMA_BF16(af, bf, c) \
    __builtin_amdgcn_wmma_f32_16x16x32_bf16(false, (af), false, (bf), (short)0, (c), false, false)

// ---------------------------------------------------------------------------
// Kernel 1: e[b] = A_b (512x512) x B_b^T, fp32 out to workspace.
// ---------------------------------------------------------------------------
__global__ __launch_bounds__(256) void k_gemm_e(const float* __restrict__ A,
                                                const float* __restrict__ B,
                                                float* __restrict__ e) {
    __shared__ __attribute__((aligned(16))) unsigned short sA[2][TM * PITCH]; // [i][d]
    __shared__ __attribute__((aligned(16))) unsigned short sB[2][TM * PITCH]; // [j][d]
    const int b = blockIdx.z, it = blockIdx.y, jt = blockIdx.x;
    const int tid = threadIdx.x, lane = tid & 31, wid = tid >> 5;
    const int wm = wid >> 2, wn = wid & 3;       // wave tile = 64(M) x 32(N)
    const int rA = tid >> 3, c4 = (tid & 7) * 4; // 128x32 float4 staging

    const float* Ab = A + ((size_t)b * SEQ + it * TM) * SEQ;
    const float* Bb = B + ((size_t)b * SEQ + jt * TM) * SEQ;

    float4 ra[4], rb[4];
    #pragma unroll
    for (int t = 0; t < 4; ++t) {
        ra[t] = *(const float4*)&Ab[(size_t)(rA + 32 * t) * SEQ + c4];
        rb[t] = *(const float4*)&Bb[(size_t)(rA + 32 * t) * SEQ + c4];
    }
    #pragma unroll
    for (int t = 0; t < 4; ++t) {
        *(uint2*)&sA[0][(rA + 32 * t) * PITCH + c4] = pack4bf(ra[t]);
        *(uint2*)&sB[0][(rA + 32 * t) * PITCH + c4] = pack4bf(rb[t]);
    }
    __syncthreads();

    v8f acc[4][2] = {};
    for (int ks = 0; ks < NSTEP; ++ks) {
        const int cur = ks & 1;
        if (ks < NSTEP - 1) {                    // prefetch next K tile into regs
            const int k0 = (ks + 1) * TK;
            #pragma unroll
            for (int t = 0; t < 4; ++t) {
                ra[t] = *(const float4*)&Ab[(size_t)(rA + 32 * t) * SEQ + k0 + c4];
                rb[t] = *(const float4*)&Bb[(size_t)(rA + 32 * t) * SEQ + k0 + c4];
            }
        }
        v16bf bf0 = load_frag_b(sB[cur], wn * 32,      lane);
        v16bf bf1 = load_frag_b(sB[cur], wn * 32 + 16, lane);
        #pragma unroll
        for (int am = 0; am < 4; ++am) {
            v16bf af = load_frag_a(sA[cur], wm * 64 + am * 16, lane);
            acc[am][0] = WMMA_BF16(af, bf0, acc[am][0]);
            acc[am][1] = WMMA_BF16(af, bf1, acc[am][1]);
        }
        if (ks < NSTEP - 1) {                    // drain regs into other buffer
            #pragma unroll
            for (int t = 0; t < 4; ++t) {
                *(uint2*)&sA[cur ^ 1][(rA + 32 * t) * PITCH + c4] = pack4bf(ra[t]);
                *(uint2*)&sB[cur ^ 1][(rA + 32 * t) * PITCH + c4] = pack4bf(rb[t]);
            }
        }
        __syncthreads();
    }

    const int half = lane >> 4, nl = lane & 15;
    #pragma unroll
    for (int am = 0; am < 4; ++am)
        #pragma unroll
        for (int bn = 0; bn < 2; ++bn)
            #pragma unroll
            for (int r = 0; r < 8; ++r) {
                int i = it * TM + wm * 64 + am * 16 + r + 8 * half;
                int j = jt * TM + wn * 32 + bn * 16 + nl;
                e[((size_t)b * SEQ + i) * SEQ + j] = acc[am][bn][r];
            }
}

// ---------------------------------------------------------------------------
// Kernel 2: per-row max & sum(exp) of e. One wave per row (wave32).
// ---------------------------------------------------------------------------
__global__ __launch_bounds__(256) void k_rowstats(const float* __restrict__ e,
                                                  float* __restrict__ rowM,
                                                  float* __restrict__ rowS) {
    const int r = blockIdx.x * 8 + (threadIdx.x >> 5);
    const int lane = threadIdx.x & 31;
    const float* p = e + (size_t)r * SEQ;
    float v[16], m = -__builtin_inff();
    #pragma unroll
    for (int t = 0; t < 16; ++t) { v[t] = p[lane + 32 * t]; m = fmaxf(m, v[t]); }
    #pragma unroll
    for (int o = 16; o >= 1; o >>= 1) m = fmaxf(m, __shfl_xor(m, o, 32));
    float s = 0.f;
    #pragma unroll
    for (int t = 0; t < 16; ++t) s += __expf(v[t] - m);
    #pragma unroll
    for (int o = 16; o >= 1; o >>= 1) s += __shfl_xor(s, o, 32);
    if (lane == 0) { rowM[r] = m; rowS[r] = s; }
}

// ---------------------------------------------------------------------------
// Kernel 3a: partial column stats over a 128-row i-chunk (online softmax).
// grid = (4 parts, 2 j-chunks, 32 batches) -> 256 workgroups.
// ---------------------------------------------------------------------------
__global__ __launch_bounds__(256) void k_colstats_part(const float* __restrict__ e,
                                                       float* __restrict__ pM,
                                                       float* __restrict__ pS) {
    const int part = blockIdx.x;
    const int b = blockIdx.z;
    const int j = blockIdx.y * 256 + threadIdx.x;
    const float* p = e + (size_t)b * SEQ * SEQ + (size_t)part * 128 * SEQ + j;
    float m = -__builtin_inff(), s = 0.f;
    for (int i = 0; i < 128; ++i) {
        float v = p[(size_t)i * SEQ];
        float mn = fmaxf(m, v);
        s = s * __expf(m - mn) + __expf(v - mn);
        m = mn;
    }
    pM[(size_t)part * BATCH * SEQ + b * SEQ + j] = m;
    pS[(size_t)part * BATCH * SEQ + b * SEQ + j] = s;
}

// ---------------------------------------------------------------------------
// Kernel 3b: combine the 4 partials per column.
// ---------------------------------------------------------------------------
__global__ __launch_bounds__(256) void k_colstats_comb(const float* __restrict__ pM,
                                                       const float* __restrict__ pS,
                                                       float* __restrict__ colM,
                                                       float* __restrict__ colS) {
    const int idx = blockIdx.x * 256 + threadIdx.x;     // 0 .. BATCH*SEQ-1
    const size_t stride = (size_t)BATCH * SEQ;
    float m = -__builtin_inff();
    #pragma unroll
    for (int p = 0; p < 4; ++p) m = fmaxf(m, pM[p * stride + idx]);
    float s = 0.f;
    #pragma unroll
    for (int p = 0; p < 4; ++p) s += pS[p * stride + idx] * __expf(pM[p * stride + idx] - m);
    colM[idx] = m;
    colS[idx] = s;
}

// ---------------------------------------------------------------------------
// Kernel 4: a_tilde = P_row @ B, fused m_a epilogue.
// ---------------------------------------------------------------------------
__global__ __launch_bounds__(256) void k_atilde(const float* __restrict__ e,
                                                const float* __restrict__ B,
                                                const float* __restrict__ Abar,
                                                const float* __restrict__ rowM,
                                                const float* __restrict__ rowS,
                                                float* __restrict__ ma) {
    __shared__ __attribute__((aligned(16))) unsigned short sP [2][TM * PITCH]; // [i][j]
    __shared__ __attribute__((aligned(16))) unsigned short sBT[2][TM * PITCH]; // [d][j]
    const int b = blockIdx.z, it = blockIdx.y, dt = blockIdx.x;
    const int tid = threadIdx.x, lane = tid & 31, wid = tid >> 5;
    const int wm = wid >> 2, wn = wid & 3;
    const int rP = tid >> 3, cP4 = (tid & 7) * 4;   // 128x32 float4 (P tile)
    const int rB = tid >> 5, cB4 = (tid & 31) * 4;  // 32x128 float4 (B tile)

    float rm[4], ri[4];
    #pragma unroll
    for (int t = 0; t < 4; ++t) {
        int gi = it * TM + rP + 32 * t;
        rm[t] = rowM[b * SEQ + gi];
        ri[t] = 1.0f / rowS[b * SEQ + gi];
    }

    float4 ra[4], rb[4];
    #pragma unroll
    for (int t = 0; t < 4; ++t) {
        int gi = it * TM + rP + 32 * t;
        ra[t] = *(const float4*)&e[((size_t)b * SEQ + gi) * SEQ + cP4];
        rb[t] = *(const float4*)&B[((size_t)b * SEQ + rB + 8 * t) * SEQ + dt * TN + cB4];
    }
    #pragma unroll
    for (int t = 0; t < 4; ++t) {
        float4 p4 = make_float4(__expf(ra[t].x - rm[t]) * ri[t], __expf(ra[t].y - rm[t]) * ri[t],
                                __expf(ra[t].z - rm[t]) * ri[t], __expf(ra[t].w - rm[t]) * ri[t]);
        *(uint2*)&sP[0][(rP + 32 * t) * PITCH + cP4] = pack4bf(p4);
        sBT[0][(cB4 + 0) * PITCH + rB + 8 * t] = f2bf1(rb[t].x);
        sBT[0][(cB4 + 1) * PITCH + rB + 8 * t] = f2bf1(rb[t].y);
        sBT[0][(cB4 + 2) * PITCH + rB + 8 * t] = f2bf1(rb[t].z);
        sBT[0][(cB4 + 3) * PITCH + rB + 8 * t] = f2bf1(rb[t].w);
    }
    __syncthreads();

    v8f acc[4][2] = {};
    for (int ks = 0; ks < NSTEP; ++ks) {
        const int cur = ks & 1;
        if (ks < NSTEP - 1) {
            const int k0 = (ks + 1) * TK;
            #pragma unroll
            for (int t = 0; t < 4; ++t) {
                int gi = it * TM + rP + 32 * t;
                ra[t] = *(const float4*)&e[((size_t)b * SEQ + gi) * SEQ + k0 + cP4];
                rb[t] = *(const float4*)&B[((size_t)b * SEQ + k0 + rB + 8 * t) * SEQ + dt * TN + cB4];
            }
        }
        v16bf bf0 = load_frag_b(sBT[cur], wn * 32,      lane);
        v16bf bf1 = load_frag_b(sBT[cur], wn * 32 + 16, lane);
        #pragma unroll
        for (int am = 0; am < 4; ++am) {
            v16bf af = load_frag_a(sP[cur], wm * 64 + am * 16, lane);
            acc[am][0] = WMMA_BF16(af, bf0, acc[am][0]);
            acc[am][1] = WMMA_BF16(af, bf1, acc[am][1]);
        }
        if (ks < NSTEP - 1) {
            #pragma unroll
            for (int t = 0; t < 4; ++t) {
                float4 p4 = make_float4(__expf(ra[t].x - rm[t]) * ri[t], __expf(ra[t].y - rm[t]) * ri[t],
                                        __expf(ra[t].z - rm[t]) * ri[t], __expf(ra[t].w - rm[t]) * ri[t]);
                *(uint2*)&sP[cur ^ 1][(rP + 32 * t) * PITCH + cP4] = pack4bf(p4);
                sBT[cur ^ 1][(cB4 + 0) * PITCH + rB + 8 * t] = f2bf1(rb[t].x);
                sBT[cur ^ 1][(cB4 + 1) * PITCH + rB + 8 * t] = f2bf1(rb[t].y);
                sBT[cur ^ 1][(cB4 + 2) * PITCH + rB + 8 * t] = f2bf1(rb[t].z);
                sBT[cur ^ 1][(cB4 + 3) * PITCH + rB + 8 * t] = f2bf1(rb[t].w);
            }
        }
        __syncthreads();
    }

    const int half = lane >> 4, nl = lane & 15;
    #pragma unroll
    for (int am = 0; am < 4; ++am)
        #pragma unroll
        for (int bn = 0; bn < 2; ++bn)
            #pragma unroll
            for (int r = 0; r < 8; ++r) {
                int i = it * TM + wm * 64 + am * 16 + r + 8 * half;
                int d = dt * TN + wn * 32 + bn * 16 + nl;
                float at = acc[am][bn][r];
                float a  = Abar[((size_t)b * SEQ + i) * SEQ + d];
                size_t base = ((size_t)b * SEQ + i) * 2048;
                ma[base + d]        = a;
                ma[base + 512 + d]  = at;
                ma[base + 1024 + d] = a - at;
                ma[base + 1536 + d] = a * at;
            }
}

// ---------------------------------------------------------------------------
// Kernel 5: b_tilde = P_col^T @ A, fused m_b epilogue.
// ---------------------------------------------------------------------------
__global__ __launch_bounds__(256) void k_btilde(const float* __restrict__ e,
                                                const float* __restrict__ Abar,
                                                const float* __restrict__ Bbar,
                                                const float* __restrict__ colM,
                                                const float* __restrict__ colS,
                                                float* __restrict__ mb) {
    __shared__ __attribute__((aligned(16))) unsigned short sPT[2][TM * PITCH]; // [j][i]
    __shared__ __attribute__((aligned(16))) unsigned short sAT[2][TM * PITCH]; // [d][i]
    const int b = blockIdx.z, jt = blockIdx.y, dt = blockIdx.x;
    const int tid = threadIdx.x, lane = tid & 31, wid = tid >> 5;
    const int wm = wid >> 2, wn = wid & 3;
    const int rB = tid >> 5, cB4 = (tid & 31) * 4;  // 32x128 float4 (both tiles)

    float cm[4], ci[4];
    #pragma unroll
    for (int q = 0; q < 4; ++q) {
        int gj = jt * TM + cB4 + q;
        cm[q] = colM[b * SEQ + gj];
        ci[q] = 1.0f / colS[b * SEQ + gj];
    }

    float4 ra[4], rb[4];
    #pragma unroll
    for (int t = 0; t < 4; ++t) {
        int gi = rB + 8 * t;
        ra[t] = *(const float4*)&e[((size_t)b * SEQ + gi) * SEQ + jt * TM + cB4];
        rb[t] = *(const float4*)&Abar[((size_t)b * SEQ + gi) * SEQ + dt * TN + cB4];
    }
    #pragma unroll
    for (int t = 0; t < 4; ++t) {
        int i = rB + 8 * t;
        sPT[0][(cB4 + 0) * PITCH + i] = f2bf1(__expf(ra[t].x - cm[0]) * ci[0]);
        sPT[0][(cB4 + 1) * PITCH + i] = f2bf1(__expf(ra[t].y - cm[1]) * ci[1]);
        sPT[0][(cB4 + 2) * PITCH + i] = f2bf1(__expf(ra[t].z - cm[2]) * ci[2]);
        sPT[0][(cB4 + 3) * PITCH + i] = f2bf1(__expf(ra[t].w - cm[3]) * ci[3]);
        sAT[0][(cB4 + 0) * PITCH + i] = f2bf1(rb[t].x);
        sAT[0][(cB4 + 1) * PITCH + i] = f2bf1(rb[t].y);
        sAT[0][(cB4 + 2) * PITCH + i] = f2bf1(rb[t].z);
        sAT[0][(cB4 + 3) * PITCH + i] = f2bf1(rb[t].w);
    }
    __syncthreads();

    v8f acc[4][2] = {};
    for (int ks = 0; ks < NSTEP; ++ks) {
        const int cur = ks & 1;
        if (ks < NSTEP - 1) {
            const int k0 = (ks + 1) * TK;
            #pragma unroll
            for (int t = 0; t < 4; ++t) {
                int gi = k0 + rB + 8 * t;
                ra[t] = *(const float4*)&e[((size_t)b * SEQ + gi) * SEQ + jt * TM + cB4];
                rb[t] = *(const float4*)&Abar[((size_t)b * SEQ + gi) * SEQ + dt * TN + cB4];
            }
        }
        v16bf bf0 = load_frag_b(sAT[cur], wn * 32,      lane);
        v16bf bf1 = load_frag_b(sAT[cur], wn * 32 + 16, lane);
        #pragma unroll
        for (int am = 0; am < 4; ++am) {
            v16bf af = load_frag_a(sPT[cur], wm * 64 + am * 16, lane);
            acc[am][0] = WMMA_BF16(af, bf0, acc[am][0]);
            acc[am][1] = WMMA_BF16(af, bf1, acc[am][1]);
        }
        if (ks < NSTEP - 1) {
            #pragma unroll
            for (int t = 0; t < 4; ++t) {
                int i = rB + 8 * t;
                sPT[cur ^ 1][(cB4 + 0) * PITCH + i] = f2bf1(__expf(ra[t].x - cm[0]) * ci[0]);
                sPT[cur ^ 1][(cB4 + 1) * PITCH + i] = f2bf1(__expf(ra[t].y - cm[1]) * ci[1]);
                sPT[cur ^ 1][(cB4 + 2) * PITCH + i] = f2bf1(__expf(ra[t].z - cm[2]) * ci[2]);
                sPT[cur ^ 1][(cB4 + 3) * PITCH + i] = f2bf1(__expf(ra[t].w - cm[3]) * ci[3]);
                sAT[cur ^ 1][(cB4 + 0) * PITCH + i] = f2bf1(rb[t].x);
                sAT[cur ^ 1][(cB4 + 1) * PITCH + i] = f2bf1(rb[t].y);
                sAT[cur ^ 1][(cB4 + 2) * PITCH + i] = f2bf1(rb[t].z);
                sAT[cur ^ 1][(cB4 + 3) * PITCH + i] = f2bf1(rb[t].w);
            }
        }
        __syncthreads();
    }

    const int half = lane >> 4, nl = lane & 15;
    #pragma unroll
    for (int am = 0; am < 4; ++am)
        #pragma unroll
        for (int bn = 0; bn < 2; ++bn)
            #pragma unroll
            for (int r = 0; r < 8; ++r) {
                int j = jt * TM + wm * 64 + am * 16 + r + 8 * half;
                int d = dt * TN + wn * 32 + bn * 16 + nl;
                float bt = acc[am][bn][r];
                float bb = Bbar[((size_t)b * SEQ + j) * SEQ + d];
                size_t base = ((size_t)b * SEQ + j) * 2048;
                mb[base + d]        = bb;
                mb[base + 512 + d]  = bt;
                mb[base + 1024 + d] = bb - bt;
                mb[base + 1536 + d] = bb * bt;
            }
}

// ---------------------------------------------------------------------------
extern "C" void kernel_launch(void* const* d_in, const int* in_sizes, int n_in,
                              void* d_out, int out_size, void* d_ws, size_t ws_size,
                              hipStream_t stream) {
    const float* A = (const float*)d_in[0];
    const float* B = (const float*)d_in[1];
    float* out = (float*)d_out;
    float* ws  = (float*)d_ws;

    const size_t E_ELEMS = (size_t)BATCH * SEQ * SEQ;   // 8,388,608 floats
    const size_t NROW    = (size_t)BATCH * SEQ;         // 16,384
    float* e    = ws;
    float* rowM = ws + E_ELEMS;
    float* rowS = rowM + NROW;
    float* colM = rowS + NROW;
    float* colS = colM + NROW;
    float* pM   = colS + NROW;          // 4 * NROW
    float* pS   = pM + 4 * NROW;        // 4 * NROW

    float* ma = out;
    float* mb = out + (size_t)BATCH * SEQ * 2048;

    dim3 gTile(SEQ / TN, SEQ / TM, BATCH);   // (4,4,32)
    k_gemm_e       <<<gTile, 256, 0, stream>>>(A, B, e);
    k_rowstats     <<<BATCH * SEQ / 8, 256, 0, stream>>>(e, rowM, rowS);
    k_colstats_part<<<dim3(4, 2, BATCH), 256, 0, stream>>>(e, pM, pS);
    k_colstats_comb<<<BATCH * SEQ / 256, 256, 0, stream>>>(pM, pS, colM, colS);
    k_atilde       <<<gTile, 256, 0, stream>>>(e, B, A, rowM, rowS, ma);
    k_btilde       <<<gTile, 256, 0, stream>>>(e, A, B, colM, colS, mb);
}
